// CostVolume_42219528520107
// MI455X (gfx1250) — compile-verified, matrix operand order checked by reading, not run on hardware
//
#include <hip/hip_runtime.h>
#include <hip/hip_bf16.h>

// ---------------------------------------------------------------------------
// CostVolume (point-cloud cost volume) for MI455X / gfx1250.
//   - All 1x1-conv MLPs run as WMMA f16 GEMMs (v_wmma_f32_16x16x32_f16),
//     f32 accumulation, intermediates stored f16 to halve HBM traffic.
//   - GEMM A-operand is register-resident across output tiles (templated on
//     K-chunk count) so each activation tensor is read from HBM exactly once.
//   - KNN: register-resident top-k (unrolled select/replace), LDS point tiles.
//   - Softmax-over-neighbors + weighted sum fused in one reducer kernel.
// ---------------------------------------------------------------------------

typedef __attribute__((ext_vector_type(16))) _Float16 v16h;
typedef __attribute__((ext_vector_type(8)))  _Float16 v8h;
typedef __attribute__((ext_vector_type(8)))  float    v8f;

#define BB 4
#define NN 4096
#define MM 8192
#define CC 64
#define KQ 32         // cross-cloud KNN
#define KS 16         // self KNN

// ------------------------------- K0: wxyz ----------------------------------
__global__ void scale_xyz_kernel(const float* __restrict__ xyz,
                                 const float* __restrict__ z,
                                 float* __restrict__ out, int total3) {
  int i = blockIdx.x * blockDim.x + threadIdx.x;
  if (i < total3) out[i] = xyz[i] * z[i / 3];
}

// ------------------------------- KNN ---------------------------------------
// One thread per query; top-KK kept in registers. Candidate points staged
// through LDS in TILE-sized chunks (bandwidth: each point read once per block).
template <int KK, int TILE>
__global__ __launch_bounds__(256)
void knn_kernel(const float* __restrict__ pts, int M,
                const float* __restrict__ qry, int Nq,
                int* __restrict__ out_idx) {
  __shared__ float sx[TILE], sy[TILE], sz[TILE];
  const int b = blockIdx.y;
  const int q = blockIdx.x * blockDim.x + threadIdx.x;
  const float* qp = qry + ((size_t)b * Nq + q) * 3;
  const float qx = qp[0], qy = qp[1], qz = qp[2];

  float bd[KK];
  int   bi[KK];
#pragma unroll
  for (int j = 0; j < KK; ++j) { bd[j] = 3.4e38f; bi[j] = 0; }
  float worst = 3.4e38f;
  int   wslot = 0;

  for (int t0 = 0; t0 < M; t0 += TILE) {
    for (int i = threadIdx.x; i < TILE; i += blockDim.x) {
      const float* pp = pts + ((size_t)b * M + t0 + i) * 3;
      sx[i] = pp[0]; sy[i] = pp[1]; sz[i] = pp[2];
    }
    __syncthreads();
    for (int i = 0; i < TILE; ++i) {
      const float dx = qx - sx[i], dy = qy - sy[i], dz = qz - sz[i];
      const float d = dx * dx + dy * dy + dz * dz;
      if (d < worst) {
        const int cand = t0 + i;
#pragma unroll
        for (int j = 0; j < KK; ++j)
          if (j == wslot) { bd[j] = d; bi[j] = cand; }
        worst = -1.f;
#pragma unroll
        for (int j = 0; j < KK; ++j)
          if (bd[j] > worst) { worst = bd[j]; wslot = j; }
      }
    }
    __syncthreads();
  }
  int* op = out_idx + ((size_t)b * Nq + q) * KK;
#pragma unroll
  for (int j = 0; j < KK; ++j) op[j] = bi[j];
}

// --------------------- K2: stage-1 feature builder --------------------------
// Row = (b,n,k). A0[96]: ch0-5 = [wxyz | qi_xyz], ch6-69 = chan-norm corr, pad 0.
__global__ __launch_bounds__(128)
void build_feat1_kernel(const float* __restrict__ wxyz,
                        const float* __restrict__ warped_points,
                        const float* __restrict__ f2_xyz,
                        const float* __restrict__ f2_points,
                        const int* __restrict__ idx1,
                        _Float16* __restrict__ A0) {
  const int wid  = threadIdx.x >> 5;
  const int lane = threadIdx.x & 31;
  const int q = blockIdx.x * 4 + wid;              // global (b,n)
  const int b = q / NN;
  const size_t row = (size_t)q * KQ + lane;
  const int nb = idx1[row];

  const float* pi = warped_points + (size_t)q * CC;
  const float* qi = f2_points + ((size_t)b * MM + nb) * CC;
  float pv[CC], qv[CC];
  float ps = 0.f, qs = 0.f;
#pragma unroll
  for (int c = 0; c < CC; ++c) { pv[c] = pi[c]; qv[c] = qi[c]; ps += pv[c]; qs += qv[c]; }
  const float pm = ps * (1.f / CC), qm = qs * (1.f / CC);
  float pss = 0.f, qss = 0.f;
#pragma unroll
  for (int c = 0; c < CC; ++c) {
    const float a = pv[c] - pm, d = qv[c] - qm;
    pss += a * a; qss += d * d;
  }
  const float pr = 1.f / fmaxf(sqrtf(pss * (1.f / (CC - 1))), 1e-12f);
  const float qr = 1.f / fmaxf(sqrtf(qss * (1.f / (CC - 1))), 1e-12f);

  _Float16* o = A0 + row * 96;
  const float* px = wxyz + (size_t)q * 3;
  const float* fx = f2_xyz + ((size_t)b * MM + nb) * 3;
  o[0] = (_Float16)px[0]; o[1] = (_Float16)px[1]; o[2] = (_Float16)px[2];
  o[3] = (_Float16)fx[0]; o[4] = (_Float16)fx[1]; o[5] = (_Float16)fx[2];
#pragma unroll
  for (int c = 0; c < CC; ++c)
    o[6 + c] = (_Float16)(((pv[c] - pm) * pr) * ((qv[c] - qm) * qr));
#pragma unroll
  for (int c = 70; c < 96; ++c) o[c] = (_Float16)0.f;
}

// --------------------- generic WMMA GEMM + bias + ReLU ----------------------
// Y[r, col0+n] = relu( sum_k X[r,k] * W[k,n] + b[n] ),  X,Y f16, W,b f32.
// Block = 128 threads (4 waves); each wave owns 16 rows.
// A (the wave's 16 x KCH*32 activation slice) is loaded from global ONCE and
// kept in registers; the cout/16 tile loop then only touches LDS for W^T.
// W^T (f16, zero-padded to cinp = 32*KCH) staged in LDS once per block.
template <int KCH>
__global__ __launch_bounds__(128)
void gemm_bias_relu_f16(const _Float16* __restrict__ X, int xpitch,
                        const float* __restrict__ W, const float* __restrict__ Bias,
                        _Float16* __restrict__ Y, int ypitch, int col0,
                        int cin, int cout) {
  __shared__ __align__(16) _Float16 sWT[128 * 192];
  __shared__ float sB[128];
  const int cinp = KCH << 5;

  for (int t = threadIdx.x; t < cout * cinp; t += blockDim.x) {
    const int nn = t / cinp, kk = t - nn * cinp;
    const float v = (kk < cin) ? W[(size_t)kk * cout + nn] : 0.f;
    sWT[nn * cinp + kk] = (_Float16)v;
  }
  for (int t = threadIdx.x; t < cout; t += blockDim.x) sB[t] = Bias[t];
  __syncthreads();

  const int lane = threadIdx.x & 31;
  const int wid  = threadIdx.x >> 5;
  const int r0   = blockIdx.x * 64 + wid * 16;
  const int m    = lane & 15;                 // A-matrix row for this lane
  const int khi  = (lane >> 4) << 3;          // 0 (lanes 0-15) or 8 (16-31)
  const int nloc = lane & 15;                 // B/D column for this lane
  const int mh   = (lane >> 4) << 3;          // D row offset per ISA layout

  // Load this wave's A slice once (per lane: KCH pairs of 16B chunks,
  // K = khi..khi+7 and khi+16..khi+23 within each 32-wide K chunk).
  v16h Areg[KCH];
  {
    const _Float16* ap = X + (size_t)(r0 + m) * xpitch + khi;
#pragma unroll
    for (int kc = 0; kc < KCH; ++kc) {
      const v8h alo = *(const v8h*)(ap + (kc << 5));
      const v8h ahi = *(const v8h*)(ap + (kc << 5) + 16);
#pragma unroll
      for (int i = 0; i < 8; ++i) { Areg[kc][i] = alo[i]; Areg[kc][i + 8] = ahi[i]; }
    }
  }

  for (int nc = 0; nc < (cout >> 4); ++nc) {
    v8f acc = {};
    const _Float16* bp0 = sWT + (size_t)(nc * 16 + nloc) * cinp + khi;
#pragma unroll
    for (int kc = 0; kc < KCH; ++kc) {
      const v8h blo = *(const v8h*)(bp0 + (kc << 5));
      const v8h bhi = *(const v8h*)(bp0 + (kc << 5) + 16);
      v16h Bv;
#pragma unroll
      for (int i = 0; i < 8; ++i) { Bv[i] = blo[i]; Bv[i + 8] = bhi[i]; }
      acc = __builtin_amdgcn_wmma_f32_16x16x32_f16(
          false, Areg[kc], false, Bv, (short)0, acc, false, false);
    }
    const float bias = sB[nc * 16 + nloc];
#pragma unroll
    for (int r = 0; r < 8; ++r) {
      float v = acc[r] + bias;
      v = v > 0.f ? v : 0.f;
      Y[(size_t)(r0 + r + mh) * ypitch + col0 + nc * 16 + nloc] = (_Float16)v;
    }
  }
}

// --------------------- K6: stage-2 feature builder --------------------------
__global__ __launch_bounds__(256)
void build_feat2_kernel(const float* __restrict__ wxyz,
                        const float* __restrict__ warped_points,
                        const float* __restrict__ feat1,
                        const int* __restrict__ idx2,
                        _Float16* __restrict__ A2,
                        _Float16* __restrict__ H2) {
  const int tid = blockIdx.x * blockDim.x + threadIdx.x;
  const int q = tid >> 4;               // (b,n)
  const int j = tid & 15;
  const int b = q / NN;
  const size_t row = (size_t)q * KS + j;
  const int nb = idx2[row];

  const float* pn = wxyz + (size_t)q * 3;
  const float* pg = wxyz + ((size_t)b * NN + nb) * 3;
  const float dx = pg[0] - pn[0], dy = pg[1] - pn[1], dz = pg[2] - pn[2];
  const float euc = sqrtf(dx * dx + dy * dy + dz * dz + 1e-20f);

  _Float16* a = A2 + row * 32;
  a[0] = (_Float16)pn[0]; a[1] = (_Float16)pn[1]; a[2] = (_Float16)pn[2];
  a[3] = (_Float16)pg[0]; a[4] = (_Float16)pg[1]; a[5] = (_Float16)pg[2];
  a[6] = (_Float16)dx; a[7] = (_Float16)dy; a[8] = (_Float16)dz;
  a[9] = (_Float16)euc;
#pragma unroll
  for (int c = 10; c < 32; ++c) a[c] = (_Float16)0.f;

  _Float16* h = H2 + row * 192;
  const float* wp = warped_points + (size_t)q * CC;
  const float* fg = feat1 + ((size_t)b * NN + nb) * CC;
#pragma unroll
  for (int c = 0; c < CC; ++c) {
    h[64 + c]  = (_Float16)wp[c];
    h[128 + c] = (_Float16)fg[c];
  }
}

// --------------------- softmax over neighbors + weighted sum ----------------
// out[q,c] = sum_k softmax_k(H[q,k,c]) * V[q,k,vcol0+c]   (64 channels)
template <int KK>
__global__ __launch_bounds__(256)
void softmax_agg_kernel(const _Float16* __restrict__ H, int hpitch,
                        const _Float16* __restrict__ V, int vpitch, int vcol0,
                        float* __restrict__ out) {
  const int tid = blockIdx.x * blockDim.x + threadIdx.x;
  const int q = tid >> 6;
  const int c = tid & 63;
  float hv[KK];
  float mx = -3.4e38f;
#pragma unroll
  for (int k = 0; k < KK; ++k) {
    hv[k] = (float)H[((size_t)q * KK + k) * hpitch + c];
    mx = fmaxf(mx, hv[k]);
  }
  float s = 0.f, acc = 0.f;
#pragma unroll
  for (int k = 0; k < KK; ++k) {
    const float e = __expf(hv[k] - mx);
    s += e;
    acc += e * (float)V[((size_t)q * KK + k) * vpitch + vcol0 + c];
  }
  out[(size_t)q * 64 + c] = acc / s;
}

// ---------------------------------------------------------------------------
static void launch_gemm(unsigned grid, const _Float16* X, int xp,
                        const float* W, const float* b,
                        _Float16* Y, int yp, int c0, int cin, int cout,
                        hipStream_t s) {
  const int kch = (cin + 31) / 32;
  switch (kch) {
    case 1: gemm_bias_relu_f16<1><<<grid, 128, 0, s>>>(X, xp, W, b, Y, yp, c0, cin, cout); break;
    case 2: gemm_bias_relu_f16<2><<<grid, 128, 0, s>>>(X, xp, W, b, Y, yp, c0, cin, cout); break;
    case 3: gemm_bias_relu_f16<3><<<grid, 128, 0, s>>>(X, xp, W, b, Y, yp, c0, cin, cout); break;
    case 4: gemm_bias_relu_f16<4><<<grid, 128, 0, s>>>(X, xp, W, b, Y, yp, c0, cin, cout); break;
    default: gemm_bias_relu_f16<6><<<grid, 128, 0, s>>>(X, xp, W, b, Y, yp, c0, cin, cout); break;
  }
}

extern "C" void kernel_launch(void* const* d_in, const int* in_sizes, int n_in,
                              void* d_out, int out_size, void* d_ws, size_t ws_size,
                              hipStream_t stream) {
  (void)in_sizes; (void)n_in; (void)out_size; (void)ws_size;
  const float* warped_xyz    = (const float*)d_in[0];
  const float* warped_points = (const float*)d_in[1];
  const float* f2_xyz        = (const float*)d_in[2];
  const float* f2_points     = (const float*)d_in[3];
  const float* lidar_z       = (const float*)d_in[4];
  const float* m1w0 = (const float*)d_in[5];
  const float* m1w1 = (const float*)d_in[6];
  const float* m1w2 = (const float*)d_in[7];
  const float* m1b0 = (const float*)d_in[8];
  const float* m1b1 = (const float*)d_in[9];
  const float* m1b2 = (const float*)d_in[10];
  const float* piw  = (const float*)d_in[11];
  const float* pib  = (const float*)d_in[12];
  const float* m2w0 = (const float*)d_in[13];
  const float* m2w1 = (const float*)d_in[14];
  const float* m2b0 = (const float*)d_in[15];
  const float* m2b1 = (const float*)d_in[16];
  const float* pcw  = (const float*)d_in[17];
  const float* pcb  = (const float*)d_in[18];
  const float* m22w0 = (const float*)d_in[19];
  const float* m22w1 = (const float*)d_in[20];
  const float* m22b0 = (const float*)d_in[21];
  const float* m22b1 = (const float*)d_in[22];

  const size_t Q  = (size_t)BB * NN;        // 16384 query points
  const size_t R0 = Q * KQ;                 // 524288 stage-1 rows
  const size_t R2 = Q * KS;                 // 262144 stage-2 rows

  // Workspace arena (aliased across stages).
  char* wsb = (char*)d_ws;
  size_t off = 0;
  auto take = [&](size_t sz) -> char* {
    char* p = wsb + off;
    off += (sz + 255) & ~(size_t)255;
    return p;
  };
  float*    wxyz  = (float*)take(Q * 3 * sizeof(float));
  int*      idx1  = (int*)take(Q * KQ * sizeof(int));
  int*      idx2  = (int*)take(Q * KS * sizeof(int));
  float*    feat1 = (float*)take(Q * CC * sizeof(float));
  _Float16* A0    = (_Float16*)take(R0 * 96 * sizeof(_Float16));   // also A2 (R2*32)
  _Float16* Y1    = (_Float16*)take(R0 * 128 * sizeof(_Float16));  // also Y4, Y6
  _Float16* Y2    = (_Float16*)take(R0 * 64 * sizeof(_Float16));   // also Y5, Y7
  _Float16* H1    = (_Float16*)take(R0 * 128 * sizeof(_Float16));  // also H2 (R2*192)
  _Float16* A2 = A0;
  _Float16* H2 = H1;

  // K0: wxyz = warped_xyz * lidar_z
  scale_xyz_kernel<<<dim3((unsigned)((Q * 3 + 255) / 256)), dim3(256), 0, stream>>>(
      warped_xyz, lidar_z, wxyz, (int)(Q * 3));

  // K1: cross-cloud KNN (unscaled warped_xyz into f2_xyz), k=32
  knn_kernel<KQ, 512><<<dim3(NN / 256, BB), dim3(256), 0, stream>>>(
      f2_xyz, MM, warped_xyz, NN, idx1);

  // K2: build A0 (xyz_cat + chan-norm corr), padded to 96 channels
  build_feat1_kernel<<<dim3((unsigned)(Q / 4)), dim3(128), 0, stream>>>(
      wxyz, warped_points, f2_xyz, f2_points, idx1, A0);

  const unsigned g1 = (unsigned)(R0 / 64);
  // mlp1: 70->128->64->64   (feat lands in H1 cols 64..127)
  launch_gemm(g1, A0, 96, m1w0, m1b0, Y1, 128, 0, 70, 128, stream);
  // pi_enc: 6->64 into H1 cols 0..63
  launch_gemm(g1, A0, 96, piw, pib, H1, 128, 0, 6, 64, stream);
  launch_gemm(g1, Y1, 128, m1w1, m1b1, Y2, 64, 0, 128, 64, stream);
  launch_gemm(g1, Y2, 64, m1w2, m1b2, H1, 128, 64, 64, 64, stream);
  // mlp2 on concat(enc, feat): 128->128->64
  launch_gemm(g1, H1, 128, m2w0, m2b0, Y1, 128, 0, 128, 128, stream);
  launch_gemm(g1, Y1, 128, m2w1, m2b1, Y2, 64, 0, 128, 64, stream);

  // K4: softmax over K=32, weighted sum of feat (H1 cols 64..127) -> feat1
  softmax_agg_kernel<KQ><<<dim3((unsigned)(Q * 64 / 256)), dim3(256), 0, stream>>>(
      Y2, 64, H1, 128, 64, feat1);

  // K5: self-KNN on scaled wxyz, k=16
  knn_kernel<KS, 512><<<dim3(NN / 256, BB), dim3(256), 0, stream>>>(
      wxyz, NN, wxyz, NN, idx2);

  // K6: build pc_cat (A2) and H2 cols 64..191 (= warped_points | gathered feat1)
  build_feat2_kernel<<<dim3((unsigned)(R2 / 256)), dim3(256), 0, stream>>>(
      wxyz, warped_points, feat1, idx2, A2, H2);

  const unsigned g2 = (unsigned)(R2 / 64);
  // pc_enc: 10->64 into H2 cols 0..63
  launch_gemm(g2, A2, 32, pcw, pcb, H2, 192, 0, 10, 64, stream);
  // mlp22: 192->128->64
  launch_gemm(g2, H2, 192, m22w0, m22b0, Y1, 128, 0, 192, 128, stream);
  launch_gemm(g2, Y1, 128, m22w1, m22b1, Y2, 64, 0, 128, 64, stream);

  // K7: softmax over ns=16, weighted sum of pc_pts_g (H2 cols 128..191) -> out
  softmax_agg_kernel<KS><<<dim3((unsigned)(Q * 64 / 256)), dim3(256), 0, stream>>>(
      Y2, 64, H2, 192, 128, (float*)d_out);
}
